// WindowMultiHeadAttentionBlock_77094662963997
// MI455X (gfx1250) — compile-verified, matrix operand order checked by reading, not run on hardware
//
#include <hip/hip_runtime.h>
#include <hip/hip_bf16.h>
#include <math.h>

// ---------------------------------------------------------------------------
// MI455X (gfx1250) implementation of a windowed MHSA transformer block.
// All large contractions run on v_wmma_f32_16x16x32_bf16 (wave32 WMMA).
// B=8,H=64,W=64,C=768, WIN=16 -> 128 windows x 256 tokens; NH=12, HD=192.
// ---------------------------------------------------------------------------

typedef __bf16 bf16_t;
typedef __attribute__((ext_vector_type(16))) __bf16 v16bf;
typedef __attribute__((ext_vector_type(8)))  float  v8f;
typedef __attribute__((ext_vector_type(4)))  unsigned int u32x4;

__device__ __forceinline__ bf16_t f2bf(float f) {
  unsigned u = __float_as_uint(f);
  unsigned r = (u + 0x7fffu + ((u >> 16) & 1u)) >> 16;
  unsigned short s = (unsigned short)r;
  return *reinterpret_cast<bf16_t*>(&s);
}

// Load one bf16 WMMA A/B fragment. For both A (16xK, lane=M-row) and B
// (Kx16 fed as Bt rows, lane=N-row), the per-lane data is two contiguous
// 8-element chunks: K = {base+off .. +7} and {base+off+16 .. +7},
// off = 8*(lane>=16).  `p` must already include base+off.
__device__ __forceinline__ v16bf load_frag(const bf16_t* p) {
  union U { u32x4 q[2]; v16bf v; } u;
  u.q[0] = *(const u32x4*)(p);
  u.q[1] = *(const u32x4*)(p + 16);
  return u.v;
}

__device__ __forceinline__ v8f wmma_bf16(v16bf a, v16bf b, v8f c) {
  return __builtin_amdgcn_wmma_f32_16x16x32_bf16(
      /*neg_a=*/false, a, /*neg_b=*/false, b,
      /*c_mod=*/(short)0, c, /*reuse_a=*/false, /*reuse_b=*/false);
}

// ---------------------------------------------------------------------------
// Weight convert + transpose: src (K x N, f32 row-major) -> dst (N x K, bf16)
// ---------------------------------------------------------------------------
__global__ void convt_kernel(const float* __restrict__ src, bf16_t* __restrict__ dst,
                             int K, int N) {
  int i = blockIdx.x * blockDim.x + threadIdx.x;
  if (i >= K * N) return;
  int k = i / N, n = i - k * N;
  dst[(long)n * K + k] = f2bf(src[i]);
}

// Gather decomposed rel-pos tables: Rh[a][k][d]=rp0[a-k+15][d], Rw likewise.
__global__ void rel_gather_kernel(const float* __restrict__ rp0,
                                  const float* __restrict__ rp1,
                                  bf16_t* __restrict__ Rh, bf16_t* __restrict__ Rw) {
  int i = blockIdx.x * blockDim.x + threadIdx.x;
  if (i >= 16 * 16 * 192) return;
  int d = i % 192;
  int rest = i / 192;
  int kk = rest & 15;
  int a = rest >> 4;
  int c = a - kk + 15;
  Rh[i] = f2bf(rp0[c * 192 + d]);
  Rw[i] = f2bf(rp1[c * 192 + d]);
}

// ---------------------------------------------------------------------------
// LayerNorm over C=768, output bf16. PART=true also performs the window
// partition: output row = window-token index (nw*256+t), source = (b,h,w).
// ---------------------------------------------------------------------------
template<bool PART>
__global__ __launch_bounds__(256) void ln_kernel(const float* __restrict__ x,
                                                 const float* __restrict__ gamma,
                                                 const float* __restrict__ beta,
                                                 bf16_t* __restrict__ out) {
  __shared__ float s1[256];
  __shared__ float s2[256];
  int row = blockIdx.x;
  long src;
  if (PART) {
    int nw = row >> 8, t = row & 255;
    int bb = nw >> 4, wi = (nw >> 2) & 3, wj = nw & 3;
    int a = t >> 4, bc = t & 15;
    src = ((long)((bb * 64 + wi * 16 + a) * 64 + wj * 16 + bc)) * 768;
  } else {
    src = (long)row * 768;
  }
  const float* xr = x + src;
  int tid = threadIdx.x;
  float s = 0.f, sq = 0.f;
  for (int c = tid; c < 768; c += 256) { float v = xr[c]; s += v; sq += v * v; }
  s1[tid] = s; s2[tid] = sq;
  __syncthreads();
  for (int st = 128; st > 0; st >>= 1) {
    if (tid < st) { s1[tid] += s1[tid + st]; s2[tid] += s2[tid + st]; }
    __syncthreads();
  }
  float mean = s1[0] * (1.f / 768.f);
  float var  = s2[0] * (1.f / 768.f) - mean * mean;
  float rstd = rsqrtf(var + 1e-5f);
  bf16_t* orow = out + (long)row * 768;
  for (int c = tid; c < 768; c += 256)
    orow[c] = f2bf((xr[c] - mean) * rstd * gamma[c] + beta[c]);
}

// ---------------------------------------------------------------------------
// Generic bf16 WMMA GEMM: C(MxN,f32) = A(MxK) @ Bt(NxK)^T + bias, with
// templated epilogue.  Block = 256 thr (8 waves); block tile 128x256;
// wave tile 64x64 (4x4 accum tiles -> 1.0 b128 fragment load per WMMA).
// M=32768; N multiple of 256; K multiple of 32.
// ---------------------------------------------------------------------------
enum { M_BF16 = 0, M_GELU = 1, M_VT = 2, M_PROJ = 3, M_RESOUT = 4 };

template<int MODE>
__global__ __launch_bounds__(256) void gemm_kernel(
    const bf16_t* __restrict__ A, int lda,
    const bf16_t* __restrict__ Bt, int K,
    const float* __restrict__ bias,
    bf16_t* __restrict__ outb, int ldo,
    float* __restrict__ outf,
    const float* __restrict__ res) {
  int tid = threadIdx.x;
  int w = tid >> 5, lane = tid & 31;
  int ln15 = lane & 15, hl = lane >> 4, off = hl << 3;
  int Mb = blockIdx.y * 128 + (w & 1) * 64;   // 2 waves along M
  int Nb = blockIdx.x * 256 + (w >> 1) * 64;  // 4 waves along N

  v8f acc[4][4] = {};
  for (int kt = 0; kt < K; kt += 32) {
    v16bf af[4];
#pragma unroll
    for (int mi = 0; mi < 4; ++mi)
      af[mi] = load_frag(A + (long)(Mb + mi * 16 + ln15) * lda + kt + off);
#pragma unroll
    for (int nj = 0; nj < 4; ++nj) {
      v16bf bfr = load_frag(Bt + (long)(Nb + nj * 16 + ln15) * K + kt + off);
#pragma unroll
      for (int mi = 0; mi < 4; ++mi)
        acc[mi][nj] = wmma_bf16(af[mi], bfr, acc[mi][nj]);
    }
  }

#pragma unroll
  for (int mi = 0; mi < 4; ++mi)
#pragma unroll
    for (int nj = 0; nj < 4; ++nj)
#pragma unroll
      for (int r = 0; r < 8; ++r) {
        int m = Mb + mi * 16 + r + (hl << 3);
        int n = Nb + nj * 16 + ln15;
        float v = acc[mi][nj][r] + bias[n];
        if (MODE == M_BF16) {
          outb[(long)m * ldo + n] = f2bf(v);
        } else if (MODE == M_GELU) {
          float g = 0.5f * v * (1.f + tanhf(0.7978845608028654f * (v + 0.044715f * v * v * v)));
          outb[(long)m * ldo + n] = f2bf(g);
        } else if (MODE == M_VT) {
          // scatter V into transposed per-head layout vt[(nw*12+h)*192+d][t]
          int t = m & 255, nw = m >> 8;
          int hh = n / 192, d = n - hh * 192;
          outb[(((long)(nw * 12 + hh)) * 192 + d) * 256 + t] = f2bf(v);
        } else if (MODE == M_PROJ) {
          // window unpartition + residual: x2 = x + attn_proj
          int nw = m >> 8, t = m & 255;
          int bb = nw >> 4, wi = (nw >> 2) & 3, wj = nw & 3;
          int a = t >> 4, bc = t & 15;
          long idx = ((long)((bb * 64 + wi * 16 + a) * 64 + wj * 16 + bc)) * 768 + n;
          outf[idx] = res[idx] + v;
        } else { // M_RESOUT: out = x2 + mlp
          long idx = (long)m * 768 + n;
          outf[idx] = res[idx] + v;
        }
      }
}

// ---------------------------------------------------------------------------
// Fused window attention per (window, head). 512 thr = 16 waves; wave w owns
// query rows [w*16, w*16+16).  Phases: rel-pos bias WMMAs -> LDS; logits
// QK^T (16 accum tiles in regs); softmax in regs (16-lane shfl reductions);
// attn@V via per-wave LDS re-fragmentation.
// ---------------------------------------------------------------------------
__global__ __launch_bounds__(512) void attn_kernel(
    const bf16_t* __restrict__ q, const bf16_t* __restrict__ k,
    const bf16_t* __restrict__ vt, const bf16_t* __restrict__ Rh,
    const bf16_t* __restrict__ Rw, bf16_t* __restrict__ attn_out) {
  __shared__ float bh[256][16];                    // bias_h[t][kh]
  __shared__ float bw[256][16];                    // bias_w[t][kw]
  __shared__ __align__(16) bf16_t stage[16][16][40]; // per-wave attn chunk

  int blk = blockIdx.x;
  int nw = blk / 12, h = blk - nw * 12;
  int tid = threadIdx.x;
  int w = tid >> 5, lane = tid & 31;
  int ln15 = lane & 15, hl = lane >> 4, off = hl << 3;

  const bf16_t* qb  = q  + (long)nw * 256 * 2304 + h * 192;
  const bf16_t* kb  = k  + (long)nw * 256 * 2304 + h * 192;
  const bf16_t* vtb = vt + ((long)(nw * 12 + h)) * 192 * 256;

  // ---- Phase 1: decomposed rel-pos bias (32 GEMMs of 16x16x192, 2/wave) ---
  for (int uu = 0; uu < 2; ++uu) {
    int u = w * 2 + uu;
    v8f bacc = {};
    const bf16_t* arow;
    const bf16_t* brow;
    if (u < 16) {          // bias_h, a = u: rows = b, cols = kh
      arow = qb + (long)(u * 16 + ln15) * 2304;
      brow = Rh + (u * 16 + ln15) * 192;
    } else {               // bias_w, b = u-16: rows = a, cols = kw
      int b = u - 16;
      arow = qb + (long)(ln15 * 16 + b) * 2304;
      brow = Rw + (b * 16 + ln15) * 192;
    }
#pragma unroll
    for (int kt = 0; kt < 192; kt += 32)
      bacc = wmma_bf16(load_frag(arow + kt + off), load_frag(brow + kt + off), bacc);
#pragma unroll
    for (int r = 0; r < 8; ++r) {
      int rr = r + (hl << 3);
      if (u < 16) bh[u * 16 + rr][ln15] = bacc[r];
      else        bw[rr * 16 + (u - 16)][ln15] = bacc[r];
    }
  }
  __syncthreads();

  // ---- Phase 2: logits = Q K^T, 16 rows x 256 cols per wave ---------------
  v8f acc[16] = {};
  {
    const bf16_t* arow = qb + (long)(w * 16 + ln15) * 2304;
    for (int kt = 0; kt < 192; kt += 32) {
      v16bf af = load_frag(arow + kt + off);
#pragma unroll
      for (int j = 0; j < 16; ++j) {
        v16bf bfr = load_frag(kb + (long)(j * 16 + ln15) * 2304 + kt + off);
        acc[j] = wmma_bf16(af, bfr, acc[j]);
      }
    }
  }

  // ---- Phase 3: scale + bias + softmax (rows live across 16-lane groups) --
  const float scale = 0.07216878364870322f; // 1/sqrt(192)
#pragma unroll
  for (int j = 0; j < 16; ++j)
#pragma unroll
    for (int r = 0; r < 8; ++r) {
      int t = w * 16 + r + (hl << 3);
      acc[j][r] = acc[j][r] * scale + bh[t][j] + bw[t][ln15];
    }
#pragma unroll
  for (int r = 0; r < 8; ++r) {
    float mx = acc[0][r];
#pragma unroll
    for (int j = 1; j < 16; ++j) mx = fmaxf(mx, acc[j][r]);
    for (int m = 1; m <= 8; m <<= 1) mx = fmaxf(mx, __shfl_xor(mx, m, 32));
    float sum = 0.f;
#pragma unroll
    for (int j = 0; j < 16; ++j) { float e = __expf(acc[j][r] - mx); acc[j][r] = e; sum += e; }
    for (int m = 1; m <= 8; m <<= 1) sum += __shfl_xor(sum, m, 32);
    float inv = 1.f / sum;
#pragma unroll
    for (int j = 0; j < 16; ++j) acc[j][r] *= inv;
  }

  // ---- Phase 4: out = attn @ V, re-fragment attn via per-wave LDS tile ----
  v8f acc2[12] = {};
  for (int kt2 = 0; kt2 < 8; ++kt2) {
#pragma unroll
    for (int jj = 0; jj < 2; ++jj) {
      int j = kt2 * 2 + jj;
#pragma unroll
      for (int r = 0; r < 8; ++r)
        stage[w][r + (hl << 3)][jj * 16 + ln15] = f2bf(acc[j][r]);
    }
    asm volatile("s_wait_dscnt 0" ::: "memory");  // wave-local LDS RAW
    v16bf af = load_frag(&stage[w][ln15][off]);
    const bf16_t* vrow0 = vtb + kt2 * 32 + off;
#pragma unroll
    for (int jd = 0; jd < 12; ++jd) {
      v16bf bfr = load_frag(vrow0 + (long)(jd * 16 + ln15) * 256);
      acc2[jd] = wmma_bf16(af, bfr, acc2[jd]);
    }
  }

  long mbase = (long)nw * 256 + w * 16;
#pragma unroll
  for (int jd = 0; jd < 12; ++jd)
#pragma unroll
    for (int r = 0; r < 8; ++r) {
      long m = mbase + r + (hl << 3);
      int n = h * 192 + jd * 16 + ln15;
      attn_out[m * 2304 + n] = f2bf(acc2[jd][r]);
    }
}

// ---------------------------------------------------------------------------
// Host-side orchestration.  Workspace plan (~835 MB), bump-allocated.
// ---------------------------------------------------------------------------
extern "C" void kernel_launch(void* const* d_in, const int* in_sizes, int n_in,
                              void* d_out, int out_size, void* d_ws, size_t ws_size,
                              hipStream_t stream) {
  const float* x   = (const float*)d_in[0];
  const float* wq  = (const float*)d_in[1];
  const float* bq  = (const float*)d_in[2];
  const float* wk  = (const float*)d_in[3];
  const float* bk  = (const float*)d_in[4];
  const float* wv  = (const float*)d_in[5];
  const float* bv  = (const float*)d_in[6];
  const float* rp0 = (const float*)d_in[7];
  const float* rp1 = (const float*)d_in[8];
  const float* wo  = (const float*)d_in[9];
  const float* bo  = (const float*)d_in[10];
  const float* g1  = (const float*)d_in[11];
  const float* o1  = (const float*)d_in[12];
  const float* g2  = (const float*)d_in[13];
  const float* o2  = (const float*)d_in[14];
  const float* w1  = (const float*)d_in[15];
  const float* b1  = (const float*)d_in[16];
  const float* w2  = (const float*)d_in[17];
  const float* b2  = (const float*)d_in[18];
  float* out = (float*)d_out;

  const long M = 32768;          // tokens (= windows*256 = B*H*W)
  char* ws = (char*)d_ws;
  size_t o = 0;
  auto alloc = [&](size_t bytes) { size_t r = o; o += (bytes + 255) & ~(size_t)255; return r; };

  bf16_t* wq_t = (bf16_t*)(ws + alloc((size_t)768 * 2304 * 2));
  bf16_t* wk_t = (bf16_t*)(ws + alloc((size_t)768 * 2304 * 2));
  bf16_t* wv_t = (bf16_t*)(ws + alloc((size_t)768 * 2304 * 2));
  bf16_t* wo_t = (bf16_t*)(ws + alloc((size_t)2304 * 768 * 2));
  bf16_t* w1_t = (bf16_t*)(ws + alloc((size_t)768 * 3072 * 2));
  bf16_t* w2_t = (bf16_t*)(ws + alloc((size_t)3072 * 768 * 2));
  bf16_t* Rh   = (bf16_t*)(ws + alloc((size_t)16 * 16 * 192 * 2));
  bf16_t* Rw   = (bf16_t*)(ws + alloc((size_t)16 * 16 * 192 * 2));
  bf16_t* xw   = (bf16_t*)(ws + alloc((size_t)M * 768 * 2));   // also reused for ln2(x2)
  bf16_t* qb   = (bf16_t*)(ws + alloc((size_t)M * 2304 * 2));
  bf16_t* kbuf = (bf16_t*)(ws + alloc((size_t)M * 2304 * 2));
  bf16_t* vtb  = (bf16_t*)(ws + alloc((size_t)M * 2304 * 2));  // (nw*12+h, d, t)
  bf16_t* aout = (bf16_t*)(ws + alloc((size_t)M * 3072 * 2));  // attn_out (M*2304) then h1 (M*3072)
  float*  x2   = (float*) (ws + alloc((size_t)M * 768 * 4));
  (void)ws_size; (void)in_sizes; (void)n_in; (void)out_size;

  // 1) weights -> bf16 transposed
  auto cgrid = [](long n) { return dim3((unsigned)((n + 255) / 256)); };
  convt_kernel<<<cgrid(768L * 2304), 256, 0, stream>>>(wq, wq_t, 768, 2304);
  convt_kernel<<<cgrid(768L * 2304), 256, 0, stream>>>(wk, wk_t, 768, 2304);
  convt_kernel<<<cgrid(768L * 2304), 256, 0, stream>>>(wv, wv_t, 768, 2304);
  convt_kernel<<<cgrid(2304L * 768), 256, 0, stream>>>(wo, wo_t, 2304, 768);
  convt_kernel<<<cgrid(768L * 3072), 256, 0, stream>>>(w1, w1_t, 768, 3072);
  convt_kernel<<<cgrid(3072L * 768), 256, 0, stream>>>(w2, w2_t, 3072, 768);
  rel_gather_kernel<<<cgrid(16L * 16 * 192), 256, 0, stream>>>(rp0, rp1, Rh, Rw);

  // 2) LN1 + window partition -> xw (bf16)
  ln_kernel<true><<<dim3((unsigned)M), 256, 0, stream>>>(x, g1, o1, xw);

  // 3) QKV projections (WMMA); block tile 128x256
  gemm_kernel<M_BF16><<<dim3(9, 256), 256, 0, stream>>>(xw, 768, wq_t, 768, bq, qb,   2304, nullptr, nullptr);
  gemm_kernel<M_BF16><<<dim3(9, 256), 256, 0, stream>>>(xw, 768, wk_t, 768, bk, kbuf, 2304, nullptr, nullptr);
  gemm_kernel<M_VT>  <<<dim3(9, 256), 256, 0, stream>>>(xw, 768, wv_t, 768, bv, vtb,  0,    nullptr, nullptr);

  // 4) fused window attention (bias + QK^T + softmax + attn@V)
  attn_kernel<<<dim3(128 * 12), 512, 0, stream>>>(qb, kbuf, vtb, Rh, Rw, aout);

  // 5) output projection + residual + window unpartition -> x2
  gemm_kernel<M_PROJ><<<dim3(3, 256), 256, 0, stream>>>(aout, 2304, wo_t, 2304, bo, nullptr, 0, x2, x);

  // 6) LN2 -> bf16 (reuse xw region)
  ln_kernel<false><<<dim3((unsigned)M), 256, 0, stream>>>(x2, g2, o2, xw);

  // 7) MLP: gelu(x @ w1 + b1) @ w2 + b2, residual from x2 -> out
  gemm_kernel<M_GELU>  <<<dim3(12, 256), 256, 0, stream>>>(xw,   768,  w1_t, 768,  b1, aout, 3072, nullptr, nullptr);
  gemm_kernel<M_RESOUT><<<dim3(3, 256),  256, 0, stream>>>(aout, 3072, w2_t, 3072, b2, nullptr, 0,  out, x2);
}
